// AlternatingHighwayLSTM_10342281248838
// MI455X (gfx1250) — compile-verified, matrix operand order checked by reading, not run on hardware
//
#include <hip/hip_runtime.h>
#include <hip/hip_bf16.h>
#include <stdint.h>

#define S_LEN 256
#define BATCH 64
#define IN0   1024
#define HID   512
#define NL    8
#define NG    2560   // 5*H
#define NPJ   3072   // 6*H
#define SB    16384  // S*B
#define LDP   56     // LDS row stride (bf16 elems): 112B = 16B-aligned, conflict-free mod 16 rows

typedef __bf16 bf16;
typedef __bf16 v16bf __attribute__((ext_vector_type(16)));
typedef __bf16 v8bf  __attribute__((ext_vector_type(8)));
typedef float  v8f   __attribute__((ext_vector_type(8)));

// ---------------------------------------------------------------- helpers
__device__ inline v8f wmma_bf16(v16bf a, v16bf b, v8f c) {
  // (neg_a, A, neg_b, B, c_mod, C, reuse_a, reuse_b)
  return __builtin_amdgcn_wmma_f32_16x16x32_bf16(false, a, false, b, (short)0, c, false, false);
}

// Async copy 16B global -> LDS (VDST = per-lane LDS byte address), ASYNCcnt-tracked.
__device__ inline void async_ld128(unsigned lds_addr, const bf16* src) {
  asm volatile("global_load_async_to_lds_b128 %0, %1, off"
               :: "v"(lds_addr), "v"(src) : "memory");
}
__device__ inline void async_wait0() {
  asm volatile("s_wait_asynccnt 0x0" ::: "memory");
}

// Build one 16x32 bf16 fragment row (A) or one column strip (B from [n][k] layout).
// 16-bit matrix VGPR layout: lanes 0-15 hold K=0..7 (v0-3) and K=16..23 (v4-7),
// lanes 16-31 hold K=8..15 and K=24..31  -> two contiguous 16B chunks per lane.
__device__ inline v16bf frag16(const bf16* rowp, int half) {
  v8bf lo = *(const v8bf*)(rowp + (half << 3));
  v8bf hi = *(const v8bf*)(rowp + 16 + (half << 3));
  v16bf r;
#pragma unroll
  for (int i = 0; i < 8; ++i) { r[i] = lo[i]; r[i + 8] = hi[i]; }
  return r;
}

__device__ inline float fsig(float x) { return 1.0f / (1.0f + __expf(-x)); }
__device__ inline float ftanh(float x) {
  x = fminf(fmaxf(x, -15.0f), 15.0f);
  float e = __expf(2.0f * x);
  return (e - 1.0f) / (e + 1.0f);
}

// Monotonic-counter grid barrier (counter memset to 0 before each layer kernel).
__device__ inline void gsync(unsigned* bar, unsigned target) {
  __syncthreads();
  if (threadIdx.x == 0) {
    __threadfence();                       // release: make stores visible at device scope
    atomicAdd(bar, 1u);
    volatile unsigned* vb = (volatile unsigned*)bar;
    while (*vb < target) __builtin_amdgcn_s_sleep(1);
  }
  __syncthreads();
  __threadfence();                         // acquire
}

// ---------------------------------------------------------------- K0: converts
__global__ void k_f32_to_bf16(const float* __restrict__ src, bf16* __restrict__ dst, long n) {
  long i = (long)blockIdx.x * blockDim.x + threadIdx.x;
  if (i < n) dst[i] = (bf16)src[i];
}

// dst[n*K + k] = src[k*N + n]  (transpose + convert: gives [n][k] bf16 layout)
__global__ void k_transpose_bf16(const float* __restrict__ src, bf16* __restrict__ dst,
                                 int K, int N) {
  long total = (long)K * N;
  long i = (long)blockIdx.x * blockDim.x + threadIdx.x;
  if (i >= total) return;
  int n = (int)(i / K), k = (int)(i % K);
  dst[i] = (bf16)src[(long)k * N + n];
}

// ---------------------------------------------------------------- K1: projection GEMM
// P[s*64+b][0..3072) = sum_k X[rev(s)][b][k] * Wx[k][n]   (X bf16, WT = Wx^T [n][k])
// Double-buffered LDS tiles filled with GLOBAL_LOAD_ASYNC_TO_LDS_B128: async engine
// copies chunk c+1 while WMMAs consume chunk c; one s_wait_asynccnt + barrier per chunk.
__global__ __launch_bounds__(256) void k_proj_gemm(
    const bf16* __restrict__ X, const bf16* __restrict__ WT,
    bf16* __restrict__ P, int K, int rev)
{
  __shared__ bf16 As[2][128 * LDP];
  __shared__ bf16 Bs[2][128 * LDP];
  const int tid = threadIdx.x;
  const int rb = blockIdx.x, cb = blockIdx.y;
  const int wave = tid >> 5, lane = tid & 31, half = lane >> 4, lr = lane & 15;
  const int wm = wave & 1, wn = wave >> 1;     // 2 x 4 wave grid: 64-row x 32-col per wave

  // staging: each thread copies 16 bf16 (32B) of one row of each tile (2 async b128 each)
  const int sr = tid >> 1;                     // tile row 0..127
  const int kc = (tid & 1) << 4;               // k sub-chunk 0 or 16
  const int gr = rb * 128 + sr;
  const int s  = gr >> 6, b = gr & 63;
  const int sn = rev ? (S_LEN - 1 - s) : s;    // fold time-reversal into the load index
  const bf16* arow = X + (long)(sn * BATCH + b) * K;
  const bf16* brow = WT + (long)(cb * 128 + sr) * K;

  // LDS byte addresses for this thread's staging slot in each buffer
  unsigned la[2], lb[2];
  la[0] = (unsigned)(size_t)&As[0][sr * LDP + kc];
  la[1] = (unsigned)(size_t)&As[1][sr * LDP + kc];
  lb[0] = (unsigned)(size_t)&Bs[0][sr * LDP + kc];
  lb[1] = (unsigned)(size_t)&Bs[1][sr * LDP + kc];

  v8f acc[4][2] = {};
  const int nc = K >> 5;

  // prologue: async-fill buffer 0 with chunk 0
  async_ld128(la[0],      arow + kc);
  async_ld128(la[0] + 16, arow + kc + 8);
  async_ld128(lb[0],      brow + kc);
  async_ld128(lb[0] + 16, brow + kc + 8);
  async_wait0();
  __syncthreads();

  for (int c = 0; c < nc; ++c) {
    if (c + 1 < nc) {                          // issue next chunk into the other buffer
      const int kn = (c + 1) << 5;
      const int nxt = (c + 1) & 1;
      async_ld128(la[nxt],      arow + kn + kc);
      async_ld128(la[nxt] + 16, arow + kn + kc + 8);
      async_ld128(lb[nxt],      brow + kn + kc);
      async_ld128(lb[nxt] + 16, brow + kn + kc + 8);
    }
    const bf16* Ab = As[c & 1];
    const bf16* Bb = Bs[c & 1];

    v16bf bfr[2];
#pragma unroll
    for (int nt = 0; nt < 2; ++nt)
      bfr[nt] = frag16(Bb + (wn * 32 + nt * 16 + lr) * LDP, half);
#pragma unroll
    for (int mt = 0; mt < 4; ++mt) {
      v16bf af = frag16(Ab + (wm * 64 + mt * 16 + lr) * LDP, half);
#pragma unroll
      for (int nt = 0; nt < 2; ++nt)
        acc[mt][nt] = wmma_bf16(af, bfr[nt], acc[mt][nt]);
    }
    async_wait0();                             // our async writes for chunk c+1 done
    __syncthreads();                           // everyone's done (reads of c, writes of c+1)
  }

  // C/D layout: lane L holds column N=L%16; VGPR r holds M = r + (L/16)*8
#pragma unroll
  for (int mt = 0; mt < 4; ++mt)
#pragma unroll
    for (int nt = 0; nt < 2; ++nt) {
      int row0 = rb * 128 + wm * 64 + mt * 16 + half * 8;
      int col  = cb * 128 + wn * 32 + nt * 16 + lr;
#pragma unroll
      for (int r = 0; r < 8; ++r)
        P[(long)(row0 + r) * NPJ + col] = (bf16)acc[mt][nt][r];
    }
}

// ---------------------------------------------------------------- K2: persistent recurrence
// 16 WGs, WG owns hidden slice j0..j0+32 (all 5 gates for that slice -> local fusion).
// Per step: z = h@Wh (WMMA, Wh^T from L2) ; z += px + bias ; gates ; h,c update.
__global__ __launch_bounds__(256) void k_lstm(
    const bf16* __restrict__ P, const bf16* __restrict__ WhT,
    const float* __restrict__ bias, const float* __restrict__ mask,
    bf16* __restrict__ hb, bf16* __restrict__ xnext, float* __restrict__ out,
    unsigned* __restrict__ bar, int rev, int last)
{
  __shared__ float zs[64][164];   // 160 z-columns (5 gates x 32) + pad
  __shared__ float cs[64][32];    // cell state lives in LDS for the whole layer
  const int tid = threadIdx.x;
  const int j0 = blockIdx.x * 32;
  const int wave = tid >> 5, lane = tid & 31, half = lane >> 4, lr = lane & 15;
  const int mt = wave & 3;                 // batch tile (4 x 16 rows)
  const int nb = (wave >> 2) * 5;          // 5 n-tiles per wave (10 total = 160 cols)
  const int jj = tid & 31;

  // init: zero c (LDS) and h ping buffer slice (global)
#pragma unroll
  for (int rep = 0; rep < 8; ++rep) {
    int b = (tid >> 5) + rep * 8;
    cs[b][jj] = 0.0f;
    hb[b * HID + j0 + jj] = (bf16)0.0f;
  }
  unsigned phase = 1;
  gsync(bar, (phase++) * gridDim.x);

  for (int t = 0; t < S_LEN; ++t) {
    const bf16* hread  = hb + (t & 1) * (BATCH * HID);
    bf16*       hwrite = hb + ((t + 1) & 1) * (BATCH * HID);

    // ---- recurrent GEMM: 64 x 160 x 512, B fragments streamed from L2-resident Wh^T
    v8f acc[5] = {};
    const bf16* arow = hread + (mt * 16 + lr) * HID;
    for (int kk = 0; kk < HID; kk += 32) {
      v16bf af = frag16(arow + kk, half);
#pragma unroll
      for (int i = 0; i < 5; ++i) {
        int ntl = nb + i;
        int col = (ntl >> 1) * HID + j0 + ((ntl & 1) << 4) + lr;  // gate*512 + j
        v16bf bfr = frag16(WhT + (long)col * HID + kk, half);
        acc[i] = wmma_bf16(af, bfr, acc[i]);
      }
    }
#pragma unroll
    for (int i = 0; i < 5; ++i) {
      int colb = (nb + i) * 16 + lr;
      int rowb = mt * 16 + half * 8;
#pragma unroll
      for (int r = 0; r < 8; ++r)
        zs[rowb + r][colb] = acc[i][r];
    }
    __syncthreads();

    // ---- gates + state update (thread owns (jj, 8 batches))
    const int j = j0 + jj;
    const int t_out = rev ? (S_LEN - 1 - t) : t;
#pragma unroll
    for (int rep = 0; rep < 8; ++rep) {
      int b = (tid >> 5) + rep * 8;
      long prow = ((long)t * BATCH + b) * NPJ;
      float zi = zs[b][jj]       + (float)P[prow + 0 * HID + j] + bias[0 * HID + j];
      float zf = zs[b][32 + jj]  + (float)P[prow + 1 * HID + j] + bias[1 * HID + j];
      float zg = zs[b][64 + jj]  + (float)P[prow + 2 * HID + j] + bias[2 * HID + j];
      float zo = zs[b][96 + jj]  + (float)P[prow + 3 * HID + j] + bias[3 * HID + j];
      float zr = zs[b][128 + jj] + (float)P[prow + 4 * HID + j] + bias[4 * HID + j];
      float lin = (float)P[prow + 5 * HID + j];
      float ig = fsig(zi), fg = fsig(zf), gg = ftanh(zg), og = fsig(zo), rg = fsig(zr);
      float c = fg * cs[b][jj] + ig * gg;
      cs[b][jj] = c;
      float h = rg * (og * ftanh(c)) + (1.0f - rg) * lin;
      hwrite[b * HID + j] = (bf16)(h * mask[b * HID + j]);   // dropout applied to recurrent path
      long orow = ((long)t_out * BATCH + b) * HID + j;       // fold output reversal
      if (last) out[orow] = h;
      else      xnext[orow] = (bf16)h;
    }
    gsync(bar, (phase++) * gridDim.x);
  }
}

// ---------------------------------------------------------------- host
extern "C" void kernel_launch(void* const* d_in, const int* in_sizes, int n_in,
                              void* d_out, int out_size, void* d_ws, size_t ws_size,
                              hipStream_t stream) {
  const float* inputs = (const float*)d_in[0];
  const float* weight = (const float*)d_in[1];
  const float* bias   = (const float*)d_in[2];
  const float* mask   = (const float*)d_in[3];
  float* out = (float*)d_out;
  char* ws = (char*)d_ws;

  unsigned* bar = (unsigned*)ws;
  bf16* xbuf = (bf16*)(ws + 256);                                   // S*B*1024 bf16
  bf16* proj = (bf16*)((char*)xbuf + (size_t)SB * IN0 * 2);         // SB*3072 bf16
  bf16* hb   = (bf16*)((char*)proj + (size_t)SB * NPJ * 2);         // 2*64*512 bf16 ping-pong
  bf16* wxT  = (bf16*)((char*)hb + (size_t)2 * BATCH * HID * 2);    // sum_l 3072*K_l bf16
  bf16* whT  = (bf16*)((char*)wxT + (size_t)(IN0 + 7 * HID) * NPJ * 2); // 8*2560*512 bf16

  { long n = (long)SB * IN0;
    k_f32_to_bf16<<<dim3((unsigned)((n + 255) / 256)), 256, 0, stream>>>(inputs, xbuf, n); }

  size_t wsrc = 0, wxo = 0;
  for (int l = 0; l < NL; ++l) {
    int K = l ? HID : IN0;
    { long n = (long)K * NPJ;
      k_transpose_bf16<<<dim3((unsigned)((n + 255) / 256)), 256, 0, stream>>>(
          weight + wsrc, wxT + wxo, K, NPJ); }
    wsrc += (size_t)K * NPJ;
    { long n = (long)HID * NG;
      k_transpose_bf16<<<dim3((unsigned)((n + 255) / 256)), 256, 0, stream>>>(
          weight + wsrc, whT + (size_t)l * NG * HID, HID, NG); }
    wsrc += (size_t)HID * NG;
    wxo += (size_t)NPJ * K;
  }

  wxo = 0;
  for (int l = 0; l < NL; ++l) {
    int K = l ? HID : IN0;
    int rev = l & 1, last = (l == NL - 1);
    dim3 g1(SB / 128, NPJ / 128);
    k_proj_gemm<<<g1, 256, 0, stream>>>(xbuf, wxT + wxo, proj, K, rev);
    wxo += (size_t)NPJ * K;
    hipMemsetAsync(bar, 0, sizeof(unsigned), stream);   // reset grid-barrier counter
    k_lstm<<<16, 256, 0, stream>>>(proj, whT + (size_t)l * NG * HID,
                                   bias + (size_t)l * NG,
                                   mask + (size_t)l * BATCH * HID,
                                   hb, xbuf, out, bar, rev, last);
  }
}
